// SAGE_2035814499042
// MI455X (gfx1250) — compile-verified, matrix operand order checked by reading, not run on hardware
//
#include <hip/hip_runtime.h>
#include <hip/hip_bf16.h>

typedef __attribute__((ext_vector_type(16))) __bf16 v16bf;
typedef __attribute__((ext_vector_type(8)))  float  v8f;

#define N_NODES 50000
#define N_EDGES 800000
#define D_IN    128
#define D_HID   128
#define D_OUT   64

// ---------------------------------------------------------------------------
// Packed fragment layouts (ISA 7.12.2, 16-bit operands, wave32):
//   A 16x32 tile: lane (half,lrow) holds row=lrow, K = k0 + {half*8+i | i<8}
//                 and k0+16+{half*8+(i-8) | i>=8}  -> 16 bf16 per lane
//   B 32x16 tile: lane (half,lcol) holds col=lcol, same K split
// Each lane's 16 elements are stored contiguously (32 B) so the GEMM does one
// 256-bit load per fragment (two coalesced global_load_b128).
__device__ __forceinline__ long packedA_index(int row, int k) {
    int rt   = row >> 4, lrow = row & 15;
    int kt   = k >> 5,   kin  = k & 31;
    int half = (kin >> 3) & 1;
    int i    = (kin & 7) + ((kin >> 4) << 3);
    int lane = half * 16 + lrow;
    return ((((long)rt * 4 + kt) * 32 + lane) << 4) + i;
}

// ---------------- utility kernels ----------------

__global__ void zero_f32(float* p, long n) {
    long i = (long)blockIdx.x * blockDim.x + threadIdx.x;
    long stride = (long)gridDim.x * blockDim.x;
    for (; i < n; i += stride) p[i] = 0.0f;
}

// f32 -> bf16 conversion into packed-A fragment layout (K = 128)
__global__ void cvt_f32_packedA(const float* __restrict__ in, __bf16* __restrict__ out, long n) {
    long i = (long)blockIdx.x * blockDim.x + threadIdx.x;
    long stride = (long)gridDim.x * blockDim.x;
    for (; i < n; i += stride) {
        int row = (int)(i >> 7);
        int k   = (int)(i & 127);
        out[packedA_index(row, k)] = (__bf16)in[i];
    }
}

// W [128 x outc] row-major f32  ->  packed-B bf16 fragments
__global__ void pack_w_bf16(const float* __restrict__ W, __bf16* __restrict__ out, int outc) {
    int tid = blockIdx.x * blockDim.x + threadIdx.x;
    int total = 128 * outc;
    if (tid >= total) return;
    int i    = tid & 15;
    int lane = (tid >> 4) & 31;
    int kt   = (tid >> 9) & 3;
    int ct   = tid >> 11;
    int half = lane >> 4;
    int col  = ct * 16 + (lane & 15);
    int k    = kt * 32 + ((i < 8) ? (half * 8 + i) : (16 + half * 8 + (i - 8)));
    out[tid] = (__bf16)W[k * outc + col];
}

// ---------------- mean aggregation ----------------

__global__ void edge_count(const int* __restrict__ dst, float* __restrict__ cnt, int e) {
    int i = blockIdx.x * blockDim.x + threadIdx.x;
    if (i < e) atomicAdd(&cnt[dst[i]], 1.0f);
}

// one thread handles one edge x 4 consecutive features (D = 128 -> 32 quads)
__global__ void scatter_sum4(const float* __restrict__ h, const int* __restrict__ src,
                             const int* __restrict__ dst, float* __restrict__ sum,
                             long total /* E*32 */) {
    long i = (long)blockIdx.x * blockDim.x + threadIdx.x;
    long stride = (long)gridDim.x * blockDim.x;
    for (; i < total; i += stride) {
        long e = i >> 5;
        int  q = (int)(i & 31);
        int  s = src[e], d = dst[e];
        const float4 v = *(const float4*)(h + (long)s * 128 + q * 4);
        float* out = sum + (long)d * 128 + q * 4;
        atomicAdd(out + 0, v.x);
        atomicAdd(out + 1, v.y);
        atomicAdd(out + 2, v.z);
        atomicAdd(out + 3, v.w);
    }
}

// out_packedA[row,k] = sum[row,k] / max(cnt[row],1)
__global__ void normalize_packedA(const float* __restrict__ sum, const float* __restrict__ cnt,
                                  __bf16* __restrict__ out, long n) {
    long i = (long)blockIdx.x * blockDim.x + threadIdx.x;
    long stride = (long)gridDim.x * blockDim.x;
    for (; i < n; i += stride) {
        int row = (int)(i >> 7);
        int k   = (int)(i & 127);
        float c = fmaxf(cnt[row], 1.0f);
        out[packedA_index(row, k)] = (__bf16)(sum[i] / c);
    }
}

// ---------------- fused dual-GEMM:  out = act(A1@B1 + A2@B2 + bias) ----------------
// K = 128, outc = NT*16. One wave32 per 16-row tile; the wave holds all NT
// 16x16 accumulators in registers (register blocking along N), so each A
// fragment is loaded exactly once per k-tile. 2*4*NT WMMAs per wave.
template <int NT>
__global__ void sage_gemm_wmma(const __bf16* __restrict__ Ap1, const __bf16* __restrict__ Ap2,
                               const __bf16* __restrict__ Bp1, const __bf16* __restrict__ Bp2,
                               const float* __restrict__ bias,
                               float* __restrict__ outF, __bf16* __restrict__ outBpacked,
                               int do_relu) {
    const int lane = threadIdx.x;        // 0..31
    const int half = lane >> 4;
    const int lrow = lane & 15;
    const int rt   = blockIdx.x;         // row tile
    const int outc = NT * 16;

    v8f c[NT];
    #pragma unroll
    for (int ct = 0; ct < NT; ++ct) {
        float bv = bias[ct * 16 + lrow];
        #pragma unroll
        for (int r = 0; r < 8; ++r) c[ct][r] = bv;
    }

    // lane's fragment bases (each kt advances 32 lanes * 16 elems = 512 bf16)
    const __bf16* a1 = Ap1 + (((long)rt * 4) * 32 + lane) * 16;
    const __bf16* a2 = Ap2 + (((long)rt * 4) * 32 + lane) * 16;
    const __bf16* b1 = Bp1 + (long)lane * 16;
    const __bf16* b2 = Bp2 + (long)lane * 16;

    #pragma unroll
    for (int kt = 0; kt < 4; ++kt) {
        const long oa = (long)kt * 512;
        v16bf a1v = *(const v16bf*)(a1 + oa);
        v16bf a2v = *(const v16bf*)(a2 + oa);
        #pragma unroll
        for (int ct = 0; ct < NT; ++ct) {
            const long ob = (long)(ct * 4 + kt) * 512;
            v16bf b1v = *(const v16bf*)(b1 + ob);
            c[ct] = __builtin_amdgcn_wmma_f32_16x16x32_bf16(false, a1v, false, b1v,
                                                            (short)0, c[ct], false, false);
            v16bf b2v = *(const v16bf*)(b2 + ob);
            c[ct] = __builtin_amdgcn_wmma_f32_16x16x32_bf16(false, a2v, false, b2v,
                                                            (short)0, c[ct], false, false);
        }
    }

    // epilogue: C/D layout -> lane<16: (M=r, N=lane); lane>=16: (M=r+8, N=lane-16)
    #pragma unroll
    for (int ct = 0; ct < NT; ++ct) {
        #pragma unroll
        for (int r = 0; r < 8; ++r) {
            const int m   = r + half * 8;
            const int row = rt * 16 + m;
            const int col = ct * 16 + lrow;
            float v = c[ct][r];
            if (do_relu) v = fmaxf(v, 0.0f);
            outF[(long)row * outc + col] = v;
            if (outBpacked) outBpacked[packedA_index(row, col)] = (__bf16)v;
        }
    }
}

// ---------------- loss ----------------

__global__ void nll_loss_kernel(const float* __restrict__ logits, const int* __restrict__ y,
                                const unsigned char* __restrict__ mask, float* __restrict__ acc,
                                int n) {
    int i = blockIdx.x * blockDim.x + threadIdx.x;
    if (i >= n) return;
    const float* row = logits + (long)i * D_OUT;
    float mx = row[0];
    #pragma unroll 4
    for (int j = 1; j < D_OUT; ++j) mx = fmaxf(mx, row[j]);
    float s = 0.0f;
    #pragma unroll 4
    for (int j = 0; j < D_OUT; ++j) s += __expf(row[j] - mx);
    float lse = mx + __logf(s);
    float nll = lse - row[y[i]];
    float m = mask[i] ? 1.0f : 0.0f;
    atomicAdd(&acc[0], nll * m);
    atomicAdd(&acc[1], m);
}

__global__ void finalize_kernel(const float* __restrict__ acc, float* __restrict__ out) {
    out[0] = acc[0] / fmaxf(acc[1], 1.0f);
}

// ---------------- launcher ----------------

extern "C" void kernel_launch(void* const* d_in, const int* in_sizes, int n_in,
                              void* d_out, int out_size, void* d_ws, size_t ws_size,
                              hipStream_t stream) {
    const float* x    = (const float*)d_in[0];
    const int*   esrc = (const int*)d_in[1];
    const int*   edst = (const int*)d_in[2];
    const int*   y    = (const int*)d_in[3];
    const unsigned char* mask = (const unsigned char*)d_in[4];  // jax bool -> 1 byte
    const float* W1l = (const float*)d_in[5];
    const float* b1  = (const float*)d_in[6];
    const float* W1r = (const float*)d_in[7];
    const float* W2l = (const float*)d_in[8];
    const float* b2  = (const float*)d_in[9];
    const float* W2r = (const float*)d_in[10];

    // workspace carve-out
    char* ws = (char*)d_ws;
    size_t off = 0;
    auto take = [&](size_t bytes) -> char* {
        off = (off + 255) & ~(size_t)255;
        char* p = ws + off;
        off += bytes;
        return p;
    };
    float*  sum_f  = (float*) take((size_t)N_NODES * 128 * 4);   // aggregation accumulator
    float*  cnt    = (float*) take((size_t)N_NODES * 4);
    __bf16* a_pk   = (__bf16*)take((size_t)N_NODES * 128 * 2);   // packed mean-aggregated feats
    __bf16* x_pk   = (__bf16*)take((size_t)N_NODES * 128 * 2);   // packed x
    float*  h_f    = (float*) take((size_t)N_NODES * 128 * 4);   // layer-1 output (f32, row-major)
    __bf16* h_pk   = (__bf16*)take((size_t)N_NODES * 128 * 2);   // packed layer-1 output
    float*  logits = (float*) take((size_t)N_NODES * D_OUT * 4);
    __bf16* w1l_pk = (__bf16*)take((size_t)128 * 128 * 2);
    __bf16* w1r_pk = (__bf16*)take((size_t)128 * 128 * 2);
    __bf16* w2l_pk = (__bf16*)take((size_t)128 * 64 * 2);
    __bf16* w2r_pk = (__bf16*)take((size_t)128 * 64 * 2);
    float*  acc    = (float*) take(2 * 4);

    const int TB = 256;
    const long nfeat = (long)N_NODES * 128;          // 6.4M
    const long nscat = (long)N_EDGES * 32;           // edge x quad
    const int  gfeat = (int)((nfeat + TB - 1) / TB);
    const int  gscat = (int)((nscat + TB - 1) / TB);
    const int  gedge = (N_EDGES + TB - 1) / TB;
    const int  gnode = (N_NODES + TB - 1) / TB;

    // weight packing + input conversion (packed fragment layouts)
    pack_w_bf16<<<64, TB, 0, stream>>>(W1l, w1l_pk, 128);
    pack_w_bf16<<<64, TB, 0, stream>>>(W1r, w1r_pk, 128);
    pack_w_bf16<<<32, TB, 0, stream>>>(W2l, w2l_pk, 64);
    pack_w_bf16<<<32, TB, 0, stream>>>(W2r, w2r_pk, 64);
    cvt_f32_packedA<<<gfeat, TB, 0, stream>>>(x, x_pk, nfeat);

    // ---- layer 1: a = mean_aggr(x) ----
    zero_f32<<<gfeat, TB, 0, stream>>>(sum_f, nfeat);
    zero_f32<<<gnode, TB, 0, stream>>>(cnt, N_NODES);
    scatter_sum4<<<gscat, TB, 0, stream>>>(x, esrc, edst, sum_f, nscat);
    edge_count<<<gedge, TB, 0, stream>>>(edst, cnt, N_EDGES);
    normalize_packedA<<<gfeat, TB, 0, stream>>>(sum_f, cnt, a_pk, nfeat);

    // h = relu(a@W1l + b1 + x@W1r)   (outc = 128 -> NT = 8)
    sage_gemm_wmma<8><<<N_NODES / 16, 32, 0, stream>>>(a_pk, x_pk, w1l_pk, w1r_pk, b1,
                                                       h_f, h_pk, 1);

    // ---- layer 2: a = mean_aggr(h) ----
    zero_f32<<<gfeat, TB, 0, stream>>>(sum_f, nfeat);
    zero_f32<<<gnode, TB, 0, stream>>>(cnt, N_NODES);
    scatter_sum4<<<gscat, TB, 0, stream>>>(h_f, esrc, edst, sum_f, nscat);
    edge_count<<<gedge, TB, 0, stream>>>(edst, cnt, N_EDGES);
    normalize_packedA<<<gfeat, TB, 0, stream>>>(sum_f, cnt, a_pk, nfeat);

    // logits = a@W2l + b2 + h@W2r   (outc = 64 -> NT = 4)
    sage_gemm_wmma<4><<<N_NODES / 16, 32, 0, stream>>>(a_pk, h_pk, w2l_pk, w2r_pk, b2,
                                                       logits, (__bf16*)nullptr, 0);

    // ---- masked NLL loss ----
    zero_f32<<<1, 32, 0, stream>>>(acc, 2);
    nll_loss_kernel<<<gnode, TB, 0, stream>>>(logits, y, mask, acc, N_NODES);
    finalize_kernel<<<1, 1, 0, stream>>>(acc, (float*)d_out);
}
